// DynamicPTHeadSelection_73280732004882
// MI455X (gfx1250) — matrix operand order repaired and checked by reading, not compiled, not run on hardware
//
#include <hip/hip_runtime.h>
#include <hip/hip_bf16.h>

typedef __bf16 bf16_t;
typedef __attribute__((ext_vector_type(16))) bf16_t v16bf;
typedef __attribute__((ext_vector_type(8)))  bf16_t v8bf;
typedef __attribute__((ext_vector_type(8)))  float  v8f;

#define B_  8
#define C_  128
#define P_  96
#define D_  128
#define H_  8
#define R_  16
#define CP_ (C_ * P_)        // 12288
#define SCALE_ 0.25f         // 1/sqrt(R)

union Frag {                 // one WMMA 16x16x32 bf16 operand (32 bytes/lane)
  v16bf    v;
  v8bf     h[2];
  unsigned u[8];
};

// pack two f32 -> packed bf16 pair (round-to-nearest-even)
__device__ __forceinline__ unsigned pk_bf16(float lo, float hi) {
  union { float f; unsigned u; } a, b; a.f = lo; b.f = hi;
  unsigned ra = a.u + 0x7FFFu + ((a.u >> 16) & 1u);
  unsigned rb = b.u + 0x7FFFu + ((b.u >> 16) & 1u);
  return (ra >> 16) | (rb & 0xFFFF0000u);
}

__device__ __forceinline__ bf16_t f2bf(float f) {
  union { float f; unsigned u; } v; v.f = f;
  unsigned r = v.u + 0x7FFFu + ((v.u >> 16) & 1u);
  unsigned short hsh = (unsigned short)(r >> 16);
  bf16_t o; __builtin_memcpy(&o, &hsh, sizeof(o));
  return o;
}

__device__ __forceinline__ v8f wmma_bf16(v16bf a, v16bf b, v8f c) {
  return __builtin_amdgcn_wmma_f32_16x16x32_bf16(false, a, false, b, (short)0, c,
                                                 false, false);
}

// ---------------------------------------------------------------------------
// Kernel 0: weights -> packed bf16.  wbf layout [sel][b][o][d], pair-packed.
// ---------------------------------------------------------------------------
__global__ __launch_bounds__(256) void wconv_kernel(
    const float* __restrict__ w0, const float* __restrict__ w1,
    const float* __restrict__ w2, const float* __restrict__ w3,
    unsigned* __restrict__ wbf) {
  const int i = blockIdx.x * 256 + threadIdx.x;    // pair idx, 4*65536 total
  const int s = i >> 16;
  const float* src = (s == 0) ? w0 : (s == 1) ? w1 : (s == 2) ? w2 : w3;
  const int off = (i & 65535) * 2;
  wbf[i] = pk_bf16(src[off], src[off + 1]);
}

// ---------------------------------------------------------------------------
// Kernel 0b: qz -> transposed bf16 copy qz_t[b][c][d][q]  (for combine GEMM B)
// ---------------------------------------------------------------------------
__global__ __launch_bounds__(256) void tconv_kernel(
    const float* __restrict__ qz, bf16_t* __restrict__ qz_t) {
  __shared__ bf16_t tile[128][106];                // padded: 53 dw stride
  const int blk = blockIdx.x;                      // b*C + c
  const size_t base = (size_t)blk * P_ * D_;
  for (int e = threadIdx.x; e < P_ * D_; e += 256) {
    const int q = e >> 7, d = e & 127;
    tile[d][q] = f2bf(qz[base + e]);
  }
  __syncthreads();
  unsigned* out32 = (unsigned*)(qz_t + (size_t)blk * D_ * P_);
  for (int e = threadIdx.x; e < D_ * P_ / 2; e += 256) {
    const int d = e / 48, q2 = (e % 48) * 2;
    union { bf16_t b[2]; unsigned u; } pp;
    pp.b[0] = tile[d][q2]; pp.b[1] = tile[d][q2 + 1];
    out32[e] = pp.u;
  }
}

// ---------------------------------------------------------------------------
// Kernel 1: projections qz[b] (12288x128) x w[b]^T -> bf16, 4 weights/block.
// A-tile staged raw (f32) via CDNA5 async global->LDS copies (ASYNCcnt).
// ---------------------------------------------------------------------------
__global__ __launch_bounds__(128) void proj_kernel(
    const float* __restrict__ qz, const bf16_t* __restrict__ wbf,
    bf16_t* __restrict__ p_tu, bf16_t* __restrict__ p_tv,
    bf16_t* __restrict__ p_cu, bf16_t* __restrict__ p_cv) {
  __shared__ float sF[16][132];                    // 528B row stride = 33x16B

  const int blk  = blockIdx.x;                     // b * 768 + mtile
  const int b    = blk / 768;
  const int mt   = blk % 768;
  const int tid  = threadIdx.x;
  const int wv   = tid >> 5;
  const int lane = tid & 31;
  const int g    = lane >> 4;
  const int ln   = lane & 15;

  const float* aRow = qz + ((size_t)b * CP_ + (size_t)mt * 16) * D_;

  // async-stage 16x128 f32 tile: one b128 async copy per row per wave
  {
#pragma unroll
    for (int i = 0; i < 4; ++i) {
      const int row = wv * 4 + i;
      unsigned ldsOff =
          (unsigned)(unsigned long long)(void*)&sF[row][0] + (unsigned)(lane * 16);
      unsigned long long ga =
          (unsigned long long)(const void*)(aRow + (size_t)row * D_) +
          (unsigned long long)(lane * 16);
      asm volatile("global_load_async_to_lds_b128 %0, %1, off"
                   :: "v"(ldsOff), "v"(ga) : "memory");
    }
    asm volatile("s_wait_asynccnt 0" ::: "memory");
  }
  __syncthreads();

  const bf16_t* wb = wbf + ((size_t)wv * B_ + b) * D_ * D_;
  bf16_t* Osel = (wv == 0) ? p_tu : (wv == 1) ? p_tv : (wv == 2) ? p_cu : p_cv;
  bf16_t* orow = Osel + ((size_t)b * CP_ + (size_t)mt * 16) * D_;

  Frag a[4];                                       // A frags: depend on kt only
#pragma unroll
  for (int kt = 0; kt < 4; ++kt) {
    const float* a0 = &sF[ln][kt * 32 + 8 * g];
    const float* a1 = &sF[ln][kt * 32 + 16 + 8 * g];
#pragma unroll
    for (int q = 0; q < 4; ++q) {
      a[kt].u[q]     = pk_bf16(a0[2 * q], a0[2 * q + 1]);
      a[kt].u[4 + q] = pk_bf16(a1[2 * q], a1[2 * q + 1]);
    }
  }

  for (int nt = 0; nt < 8; ++nt) {
    const int n = nt * 16 + ln;
    __builtin_prefetch(wb + (size_t)n * D_, 0, 3);
    v8f acc = {};
#pragma unroll
    for (int kt = 0; kt < 4; ++kt) {
      Frag bb;
      const bf16_t* wp = wb + (size_t)n * D_ + kt * 32 + g * 8;
      bb.h[0] = *(const v8bf*)wp;
      bb.h[1] = *(const v8bf*)(wp + 16);
      acc = wmma_bf16(a[kt].v, bb.v, acc);
    }
#pragma unroll
    for (int r = 0; r < 8; ++r)
      orow[(size_t)(r + (g << 3)) * D_ + nt * 16 + ln] = f2bf(acc[r]);
  }
}

// ---------------------------------------------------------------------------
// Kernel 2: time attention, one block per (b,c); 6 waves own 16-row strips.
// ---------------------------------------------------------------------------
__global__ __launch_bounds__(192) void time_attn_kernel(
    const float* __restrict__ qz, const bf16_t* __restrict__ qz_t,
    const bf16_t* __restrict__ p_tu, const bf16_t* __restrict__ p_tv,
    float* __restrict__ out_t) {
  __shared__ float sAbar[96][100];                 // 400B stride = 25x16B

  const int blk  = blockIdx.x;                     // b*C + c
  const int b    = blk >> 7;
  const int c    = blk & 127;
  const int tid  = threadIdx.x;
  const int wv   = tid >> 5;                       // 0..5
  const int lane = tid & 31;
  const int g    = lane >> 4;
  const int ln   = lane & 15;

  const size_t rowbase = ((size_t)b * CP_ + (size_t)c * P_) * D_;
  const bf16_t* U = p_tu + rowbase;
  const bf16_t* V = p_tv + rowbase;
  v8f vz = {};

  v8f macc[6];
#pragma unroll
  for (int t = 0; t < 6; ++t) macc[t] = vz;

  for (int h = 0; h < H_; ++h) {
    Frag a;                                        // real K=16: half zero-padded
    a.h[0] = *(const v8bf*)(U + (size_t)(wv * 16 + ln) * D_ + h * R_);
    a.h[1] = *(const v8bf*)(U + (size_t)(wv * 16 + ln) * D_ + h * R_ + 8);

    v8f lg[6];
#pragma unroll
    for (int t = 0; t < 6; ++t) {
      Frag bb;
      if (g == 0) {
        const bf16_t* vp = V + (size_t)(t * 16 + ln) * D_ + h * R_;
        bb.h[0] = *(const v8bf*)vp;
        bb.h[1] = *(const v8bf*)(vp + 8);
      } else {
#pragma unroll
        for (int q = 0; q < 8; ++q) bb.u[q] = 0;
      }
      lg[t] = wmma_bf16(a.v, bb.v, vz);
    }
#pragma unroll
    for (int r = 0; r < 8; ++r) {                  // row softmax over 96
      float x[6]; float mx = -3.0e38f;
#pragma unroll
      for (int t = 0; t < 6; ++t) { x[t] = lg[t][r]; mx = fmaxf(mx, x[t]); }
#pragma unroll
      for (int m = 1; m <= 8; m <<= 1) mx = fmaxf(mx, __shfl_xor(mx, m, 32));
      float e[6]; float s = 0.f;
#pragma unroll
      for (int t = 0; t < 6; ++t) { e[t] = __expf((x[t] - mx) * SCALE_); s += e[t]; }
#pragma unroll
      for (int m = 1; m <= 8; m <<= 1) s += __shfl_xor(s, m, 32);
      const float inv = 0.125f / s;                // head mean folded in
#pragma unroll
      for (int t = 0; t < 6; ++t) macc[t][r] += e[t] * inv;
    }
  }

#pragma unroll
  for (int t = 0; t < 6; ++t)
#pragma unroll
    for (int r = 0; r < 8; ++r)
      sAbar[wv * 16 + r + (g << 3)][t * 16 + ln] = macc[t][r];
  __syncthreads();

  // combine: out_t[b,c] = Abar (96x96) x qz[b,c] (96x128), B from qz_t (bf16)
  Frag af[3];                                      // A frags depend on kt only
  const float* arow = &sAbar[wv * 16 + ln][0];
#pragma unroll
  for (int kt = 0; kt < 3; ++kt) {
    const float* a0 = arow + kt * 32 + 8 * g;
    const float* a1 = arow + kt * 32 + 16 + 8 * g;
#pragma unroll
    for (int q = 0; q < 4; ++q) {
      af[kt].u[q]     = pk_bf16(a0[2 * q], a0[2 * q + 1]);
      af[kt].u[4 + q] = pk_bf16(a1[2 * q], a1[2 * q + 1]);
    }
  }
  for (int nt = 0; nt < 8; ++nt) {
    const bf16_t* bp = qz_t + ((size_t)(b * C_ + c) * D_ + nt * 16 + ln) * P_;
    v8f acc = {};
#pragma unroll
    for (int kt = 0; kt < 3; ++kt) {
      Frag bb;
      bb.h[0] = *(const v8bf*)(bp + kt * 32 + 8 * g);
      bb.h[1] = *(const v8bf*)(bp + kt * 32 + 16 + 8 * g);
      acc = wmma_bf16(af[kt].v, bb.v, acc);
    }
#pragma unroll
    for (int r = 0; r < 8; ++r)
      out_t[rowbase + (size_t)(wv * 16 + r + (g << 3)) * D_ + nt * 16 + ln] = acc[r];
  }
}

// ---------------------------------------------------------------------------
// Kernel 3: channel attention, one block per (b,p); 8 waves own row strips.
// ---------------------------------------------------------------------------
__global__ __launch_bounds__(256) void chan_attn_kernel(
    const float* __restrict__ qz,
    const bf16_t* __restrict__ p_cu, const bf16_t* __restrict__ p_cv,
    float* __restrict__ out_c) {
  const int blk  = blockIdx.x;                     // b*P + p
  const int b    = blk / P_;
  const int p    = blk % P_;
  const int tid  = threadIdx.x;
  const int wv   = tid >> 5;                       // 0..7
  const int lane = tid & 31;
  const int g    = lane >> 4;
  const int ln   = lane & 15;

  v8f vz = {};
  v8f macc[8];
#pragma unroll
  for (int t = 0; t < 8; ++t) macc[t] = vz;

  const size_t abase = (((size_t)b * C_ + (wv * 16 + ln)) * P_ + p) * D_;

  for (int h = 0; h < H_; ++h) {
    Frag a;
    a.h[0] = *(const v8bf*)(p_cu + abase + h * R_);
    a.h[1] = *(const v8bf*)(p_cu + abase + h * R_ + 8);

    v8f lg[8];
#pragma unroll
    for (int t = 0; t < 8; ++t) {
      Frag bb;
      if (g == 0) {
        const bf16_t* vp =
            p_cv + (((size_t)b * C_ + (t * 16 + ln)) * P_ + p) * D_ + h * R_;
        bb.h[0] = *(const v8bf*)vp;
        bb.h[1] = *(const v8bf*)(vp + 8);
      } else {
#pragma unroll
        for (int q = 0; q < 8; ++q) bb.u[q] = 0;
      }
      lg[t] = wmma_bf16(a.v, bb.v, vz);
    }
#pragma unroll
    for (int r = 0; r < 8; ++r) {                  // softmax over d (128)
      float x[8]; float mx = -3.0e38f;
#pragma unroll
      for (int t = 0; t < 8; ++t) { x[t] = lg[t][r]; mx = fmaxf(mx, x[t]); }
#pragma unroll
      for (int m = 1; m <= 8; m <<= 1) mx = fmaxf(mx, __shfl_xor(mx, m, 32));
      float e[8]; float s = 0.f;
#pragma unroll
      for (int t = 0; t < 8; ++t) { e[t] = __expf((x[t] - mx) * SCALE_); s += e[t]; }
#pragma unroll
      for (int m = 1; m <= 8; m <<= 1) s += __shfl_xor(s, m, 32);
      const float inv = 0.125f / s;
#pragma unroll
      for (int t = 0; t < 8; ++t) macc[t][r] += e[t] * inv;
    }
  }

  // out[b, c, p, d] = mean-attn[c, d] * qz[b, c, p, d]
#pragma unroll
  for (int t = 0; t < 8; ++t)
#pragma unroll
    for (int r = 0; r < 8; ++r) {
      const int cc = wv * 16 + r + (g << 3);
      const int d  = t * 16 + ln;
      const size_t idx = (((size_t)b * C_ + cc) * P_ + p) * D_ + d;
      out_c[idx] = macc[t][r] * qz[idx];
    }
}

// ---------------------------------------------------------------------------
extern "C" void kernel_launch(void* const* d_in, const int* in_sizes, int n_in,
                              void* d_out, int out_size, void* d_ws, size_t ws_size,
                              hipStream_t stream) {
  (void)in_sizes; (void)n_in; (void)out_size; (void)ws_size;
  const float* qz  = (const float*)d_in[0];
  const float* wtu = (const float*)d_in[1];
  const float* wtv = (const float*)d_in[2];
  const float* wcu = (const float*)d_in[3];
  const float* wcv = (const float*)d_in[4];

  const size_t projElems = (size_t)B_ * CP_ * D_;       // 12,582,912
  bf16_t* p_tu = (bf16_t*)d_ws;
  bf16_t* p_tv = p_tu + projElems;
  bf16_t* p_cu = p_tv + projElems;
  bf16_t* p_cv = p_cu + projElems;
  bf16_t* qz_t = p_cv + projElems;                      // [b][c][d][q]
  bf16_t* wbf  = qz_t + projElems;                      // 4x[b][o][d]

  float* out_t = (float*)d_out;
  float* out_c = out_t + projElems;

  wconv_kernel<<<dim3(1024), dim3(256), 0, stream>>>(
      wtu, wtv, wcu, wcv, (unsigned*)wbf);
  tconv_kernel<<<dim3(B_ * C_), dim3(256), 0, stream>>>(qz, qz_t);
  proj_kernel<<<dim3(B_ * 768), dim3(128), 0, stream>>>(
      qz, wbf, p_tu, p_tv, p_cu, p_cv);
  time_attn_kernel<<<dim3(B_ * C_), dim3(192), 0, stream>>>(
      qz, qz_t, p_tu, p_tv, out_t);
  chan_attn_kernel<<<dim3(B_ * P_), dim3(256), 0, stream>>>(
      qz, p_cu, p_cv, out_c);
}
